// ShapeDynamicFeedForwardSecond_36266703847580
// MI455X (gfx1250) — matrix operand order repaired and checked by reading, not compiled, not run on hardware
//
#include <hip/hip_runtime.h>
#include <stdint.h>

// ---------------------------------------------------------------------------
// MI455X (gfx1250) dynamic 3D conv, implicit GEMM on v_wmma_f32_16x16x32_bf16
// A-tiles staged into LDS by the Tensor Data Mover (TDM) with row padding.
// ---------------------------------------------------------------------------
typedef __attribute__((ext_vector_type(16))) __bf16 v16bf;
typedef __attribute__((ext_vector_type(8)))  float  v8f;
typedef __attribute__((ext_vector_type(4)))  unsigned int u32x4;
typedef __attribute__((ext_vector_type(8)))  int i32x8;
typedef __attribute__((ext_vector_type(4)))  int i32x4;

#define N_    8
#define CIN   32
#define COUT  32
#define T_    16
#define H_    112
#define W_    112
#define THW   (T_*H_*W_)            // 200704
#define KS    27                    // 3*3*3 taps
#define EPS_  1e-5f

// workspace float layout
#define WS_PHI    0                 // 64 floats
#define WS_DYNB   64                // 256 floats
#define WS_SUM    320               // 32
#define WS_SQ     352               // 32
#define WS_SCALE  384               // 32
#define WS_SHIFT  416               // 32
#define WS_ABUF_BYTES 2048          // bf16 A tiles: [b][tap][c][i], 8*27*32*32*2 B
#define ABYTES_G  (KS*COUT*CIN*2)   // 55296 B per sample in global (dense 64B rows)

// conv-kernel LDS layout (dynamic)
#define IPAD   40                   // i-dim padded to 40 elems = 80 B
#define XW     114                  // 112 + halo
#define XBYTES (9*XW*IPAD*2)        // 82080
#define AROW   80                   // padded A row: 64 B data + 16 B pad (bank-friendly)
#define ASTEP  (COUT*AROW)          // 2560 B per tap
#define ABYTES_L (KS*ASTEP)         // 69120
#define SMEM_TOTAL (XBYTES + ABYTES_L + 256)

__device__ __forceinline__ uint16_t f2bf(float f) {
  uint32_t u = __float_as_uint(f);
  uint32_t r = u + 0x7FFFu + ((u >> 16) & 1u);   // round-to-nearest-even
  return (uint16_t)(r >> 16);
}

// --------------------------------------------------------------------- zero
__global__ void k_zero_stats(float* ws) {
  int i = threadIdx.x;
  if (i < 64) ws[WS_SUM + i] = 0.f;
}

// ------------------------------------------------------------------ routing
__global__ void k_routing(const float* __restrict__ stdx, const int* __restrict__ epochs,
                          const float* __restrict__ w1, const float* __restrict__ b1,
                          const float* __restrict__ w2, const float* __restrict__ b2,
                          const float* __restrict__ fcw, const float* __restrict__ fcb,
                          float* __restrict__ ws) {
  __shared__ float sx[32][32];
  __shared__ float c1[16][28][28];
  __shared__ int   hmax[32];
  __shared__ float logits[8];
  const int b = blockIdx.x, tid = threadIdx.x;
  for (int i = tid; i < 1024; i += blockDim.x) sx[i >> 5][i & 31] = stdx[b*1024 + i];
  if (tid < 32) hmax[tid] = 0;                    // relu output >= 0
  __syncthreads();
  for (int idx = tid; idx < 16*28*28; idx += blockDim.x) {
    int oc = idx / 784, r = idx % 784, oh = r / 28, ow = r % 28;
    float acc = b1[oc];
    for (int kh = 0; kh < 5; ++kh)
      for (int kw = 0; kw < 5; ++kw)
        acc = fmaf(sx[oh+kh][ow+kw], w1[oc*25 + kh*5 + kw], acc);
    c1[oc][oh][ow] = fmaxf(acc, 0.f);
  }
  __syncthreads();
  for (int idx = tid; idx < 32*24*24; idx += blockDim.x) {
    int oc = idx / 576, r = idx % 576, oh = r / 24, ow = r % 24;
    float acc = b2[oc];
    for (int ic = 0; ic < 16; ++ic)
      for (int kh = 0; kh < 5; ++kh)
        for (int kw = 0; kw < 5; ++kw)
          acc = fmaf(c1[ic][oh+kh][ow+kw], w2[((oc*16+ic)*5+kh)*5 + kw], acc);
    atomicMax(&hmax[oc], __float_as_int(fmaxf(acc, 0.f)));
  }
  __syncthreads();
  if (tid < 8) {
    float acc = fcb[tid];
    for (int c = 0; c < 32; ++c) acc = fmaf(__int_as_float(hmax[c]), fcw[tid*32 + c], acc);
    logits[tid] = acc;
  }
  __syncthreads();
  if (tid == 0) {
    int e = epochs[0];
    float tau = (e < 10) ? (30.0f - 2.9f * (float)e) : 1.0f;
    float mx = -1e30f;
    for (int n = 0; n < 8; ++n) mx = fmaxf(mx, logits[n] / tau);
    float s = 0.f, ex[8];
    for (int n = 0; n < 8; ++n) { ex[n] = __expf(logits[n] / tau - mx); s += ex[n]; }
    for (int n = 0; n < 8; ++n) ws[WS_PHI + b*8 + n] = ex[n] / s;
  }
}

// ------------------------------------------------- filter mixing -> bf16 A
// Global A layout per (b, tap): [c][i] dense row-major bf16 (64 B per c-row)
__global__ void k_mix(const float* __restrict__ weights, const float* __restrict__ biases,
                      float* __restrict__ ws) {
  const int b = blockIdx.x / KS, s = blockIdx.x % KS;
  float phi[8];
  for (int n = 0; n < 8; ++n) phi[n] = ws[WS_PHI + b*8 + n];
  uint16_t* abuf = (uint16_t*)((char*)ws + WS_ABUF_BYTES);
  for (int e = threadIdx.x; e < COUT*CIN; e += blockDim.x) {
    int c = e >> 5, i = e & 31;
    float acc = 0.f;
    for (int n = 0; n < 8; ++n)
      acc = fmaf(phi[n], weights[(n*COUT + c)*(CIN*KS) + i*KS + s], acc);
    abuf[(b*KS + s)*(COUT*CIN) + e] = f2bf(acc);
  }
  if (s == 0 && threadIdx.x < COUT) {
    float acc = 0.f;
    for (int n = 0; n < 8; ++n) acc = fmaf(phi[n], biases[n*COUT + threadIdx.x], acc);
    ws[WS_DYNB + b*COUT + threadIdx.x] = acc;
  }
}

// ---------------------------------------------------- TDM descriptor helper
// 2D tile: 864 rows x 32 bf16 (64 B) from global, +16 B LDS pad per row.
__device__ __forceinline__ void tdm_load_A(const void* gsrc, uint32_t lds_off) {
#if defined(__gfx1250__) && __has_builtin(__builtin_amdgcn_tensor_load_to_lds)
  uint64_t ga = (uint64_t)(uintptr_t)gsrc;
  u32x4 g0;
  g0[0] = 1u;                                   // count=1 (valid), user mode
  g0[1] = lds_off;                              // lds_addr (bytes)
  g0[2] = (uint32_t)ga;                         // global_addr[95:64]
  g0[3] = (uint32_t)((ga >> 32) & 0x01FFFFFFu)  // global_addr[120:96]
        | (2u << 30);                           // type=2 ("image")
  i32x8 g1;
  g1[0] = (1 << 16)                             // data_size = 1 -> 2 bytes
        | (1 << 20)                             // pad_enable
        | (3 << 22)                             // pad_interval: 16 DWORDs (64 B)
        | (3 << 25);                            // pad_amount: 4 DWORDs (16 B)
  g1[1] = 0;                                    // no atomic barrier
  g1[2] = (CIN & 0xFFFF) << 16;                 // tensor_dim0 = 32 (lo16 @ bits 63:48)
  g1[3] = ((KS*COUT) & 0xFFFF) << 16;           // tensor_dim1 = 864 (lo16 @ 111:96)... 
  // careful: tensor_dim0 hi16 @ bits 79:64 (g1[2] lo of next dword) is 0;
  // layout: d2 = dim0_hi16 | dim1_lo16<<16 ; d3 = dim1_hi16 | tile_dim0<<16
  g1[2] = ((KS*COUT) & 0xFFFF) << 16;           // d2: dim0_hi=0 | dim1_lo=864
  g1[3] = (CIN & 0xFFFF) << 16;                 // d3: dim1_hi=0 | tile_dim0=32
  g1[4] = (KS*COUT) & 0xFFFF;                   // d4: tile_dim1=864 | tile_dim2=0
  g1[5] = CIN;                                  // d5: tensor_dim0_stride lo32 = 32
  g1[6] = 0;                                    // d6: stride0_hi | stride1_lo
  g1[7] = 0;
  i32x4 z4 = {0, 0, 0, 0};
#if __clang_major__ >= 23
  i32x8 z8 = {0, 0, 0, 0, 0, 0, 0, 0};
  __builtin_amdgcn_tensor_load_to_lds(g0, g1, z4, z4, z8, 0);
#else
  __builtin_amdgcn_tensor_load_to_lds(g0, g1, z4, z4, 0);
#endif
  __builtin_amdgcn_s_wait_tensorcnt(0);
#else
  (void)gsrc; (void)lds_off;
#endif
}

// ------------------------------------------------------------ WMMA conv
// One block per output row (b, t, h): M=32 channels (2 m-tiles), N=112 (7 waves
// x 16), K = 27 taps x 32 input channels (27 WMMA K-steps of 32).
__global__ void __launch_bounds__(224)
k_conv(const float* __restrict__ x, float* __restrict__ out, float* __restrict__ ws) {
  extern __shared__ char smem[];
  uint16_t* xl = (uint16_t*)smem;                       // [tap(kd*3+kh)][w'][IPAD]
  char*     al = smem + XBYTES;                         // [tap][c][AROW bytes]
  float* redsum = (float*)(smem + XBYTES + ABYTES_L);   // [32]
  float* redsq  = redsum + 32;                          // [32]

  const int b = blockIdx.y;
  const int t = blockIdx.x / H_;
  const int h = blockIdx.x % H_;
  const int tid = threadIdx.x;

  if (tid < 32) { redsum[tid] = 0.f; redsq[tid] = 0.f; }

  const char* agbl = (const char*)ws + WS_ABUF_BYTES + (size_t)b * ABYTES_G;
#if defined(__gfx1250__) && __has_builtin(__builtin_amdgcn_tensor_load_to_lds)
  // TDM: one wave issues the whole 864x64B tile copy with 16B/row LDS padding.
  if (tid < 32) tdm_load_A(agbl, (uint32_t)(uintptr_t)al);
#else
  // manual fallback: dword copy with row padding
  for (int d = tid; d < KS*COUT*16; d += 224) {
    int r = d >> 4, q = d & 15;
    ((uint32_t*)al)[r*20 + q] = ((const uint32_t*)agbl)[d];
  }
#endif

  // stage x halo rows (9 taps x 32 ch x 114 cols) as bf16, i innermost
  for (int idx = tid; idx < 9*CIN*XW; idx += 224) {
    int kdkh = idx / (CIN*XW);
    int r    = idx - kdkh*(CIN*XW);
    int i    = r / XW;
    int wp   = r - i*XW;
    int kd = kdkh / 3, kh = kdkh % 3;
    int tt = t + kd - 1, hh = h + kh - 1, wc = wp - 1;
    float v = 0.f;
    if (tt >= 0 && tt < T_ && hh >= 0 && hh < H_ && wc >= 0 && wc < W_)
      v = x[(((size_t)(b*CIN + i)*T_ + tt)*H_ + hh)*W_ + wc];
    xl[(kdkh*XW + wp)*IPAD + i] = f2bf(v);
  }
  __syncthreads();

  const int wave = tid >> 5;          // 0..6 -> n-tile
  const int lane = tid & 31;
  const int nsub = lane & 15;
  const int half = lane >> 4;
  const int p    = wave*16 + nsub;    // output column 0..111

  v8f acc0 = {}; v8f acc1 = {};
  const char* xbase = (const char*)xl + (size_t)p * (IPAD*2) + half*32;
  const char* abase0 = al + nsub*AROW + half*32;
  const char* abase1 = al + (16 + nsub)*AROW + half*32;

  #pragma unroll 1
  for (int kdkh = 0; kdkh < 9; ++kdkh) {
    const char* xrow = xbase + (size_t)kdkh * (XW*IPAD*2);
    const char* arow0 = abase0 + kdkh*3*ASTEP;
    const char* arow1 = abase1 + kdkh*3*ASTEP;
    #pragma unroll
    for (int kw = 0; kw < 3; ++kw) {
      // B fragment (32x16 bf16): lane(l) col=l%16, K(i) = half*16 + 2v,2v+1
      union { uint4 q[2]; v16bf v; } B;
      const char* brow = xrow + kw*(IPAD*2);
      B.q[0] = *(const uint4*)(brow);
      B.q[1] = *(const uint4*)(brow + 16);
      // A fragments (16x32 bf16): lane row m=l%16, K = half*16 + 2vi
      union { uint4 q[2]; v16bf v; } A0, A1;
      const char* a0 = arow0 + kw*ASTEP;
      A0.q[0] = *(const uint4*)(a0);
      A0.q[1] = *(const uint4*)(a0 + 16);
      const char* a1 = arow1 + kw*ASTEP;
      A1.q[0] = *(const uint4*)(a1);
      A1.q[1] = *(const uint4*)(a1 + 16);

      acc0 = __builtin_amdgcn_wmma_f32_16x16x32_bf16(false, A0.v, false, B.v,
                                                     (short)0, acc0, false, false);
      acc1 = __builtin_amdgcn_wmma_f32_16x16x32_bf16(false, A1.v, false, B.v,
                                                     (short)0, acc1, false, false);
    }
  }

  // epilogue: + dyn bias, store pre-BN y, fused BN partial stats
  const float* dynb = ws + WS_DYNB + b*COUT;
  float* outp = out + (size_t)b*COUT*THW + (size_t)t*(H_*W_) + h*W_ + p;
  for (int mt = 0; mt < 2; ++mt) {
    v8f a = mt ? acc1 : acc0;
    for (int v = 0; v < 8; ++v) {
      int c = mt*16 + half*8 + v;              // C/D layout: VGPR v -> M = half*8+v
      float val = a[v] + dynb[c];
      outp[(size_t)c * THW] = val;
      float s1 = val, s2 = val * val;          // reduce over the 16 n-lanes
      for (int m = 8; m >= 1; m >>= 1) {
        s1 += __shfl_xor(s1, m, 32);
        s2 += __shfl_xor(s2, m, 32);
      }
      if (nsub == 0) { atomicAdd(&redsum[c], s1); atomicAdd(&redsq[c], s2); }
    }
  }
  __syncthreads();
  if (tid < 32) {
    atomicAdd(&ws[WS_SUM + tid], redsum[tid]);
    atomicAdd(&ws[WS_SQ  + tid], redsq[tid]);
  }
}

// --------------------------------------------------------------- finalize
__global__ void k_finalize(const float* __restrict__ gamma, const float* __restrict__ beta,
                           float* ws) {
  int c = threadIdx.x;
  if (c < 32) {
    const float M = (float)N_ * (float)THW;    // 1605632
    float mean = ws[WS_SUM + c] / M;
    float var  = ws[WS_SQ  + c] / M - mean * mean;
    float inv  = rsqrtf(var + EPS_);
    float sc   = gamma[c] * inv;
    ws[WS_SCALE + c] = sc;
    ws[WS_SHIFT + c] = beta[c] - mean * sc;
  }
}

// ------------------------------------------------------------- BN + ReLU
__global__ void k_bnrelu(float* __restrict__ out, const float* __restrict__ ws) {
  const size_t total4 = (size_t)N_ * COUT * THW / 4;
  const size_t stride = (size_t)gridDim.x * blockDim.x;
  for (size_t i = (size_t)blockIdx.x * blockDim.x + threadIdx.x; i < total4; i += stride) {
    float4 v = ((float4*)out)[i];
    int c = (int)((i * 4 / (size_t)THW) % COUT);
    float sc = ws[WS_SCALE + c], sh = ws[WS_SHIFT + c];
    v.x = fmaxf(fmaf(v.x, sc, sh), 0.f);
    v.y = fmaxf(fmaf(v.y, sc, sh), 0.f);
    v.z = fmaxf(fmaf(v.z, sc, sh), 0.f);
    v.w = fmaxf(fmaf(v.w, sc, sh), 0.f);
    ((float4*)out)[i] = v;
  }
}

// ---------------------------------------------------------------- launch
extern "C" void kernel_launch(void* const* d_in, const int* in_sizes, int n_in,
                              void* d_out, int out_size, void* d_ws, size_t ws_size,
                              hipStream_t stream) {
  (void)in_sizes; (void)n_in; (void)out_size; (void)ws_size;
  const float* x      = (const float*)d_in[0];
  const float* stdx   = (const float*)d_in[1];
  const int*   epochs = (const int*)  d_in[2];
  const float* wts    = (const float*)d_in[3];
  const float* biases = (const float*)d_in[4];
  const float* c1w    = (const float*)d_in[5];
  const float* c1b    = (const float*)d_in[6];
  const float* c2w    = (const float*)d_in[7];
  const float* c2b    = (const float*)d_in[8];
  const float* fcw    = (const float*)d_in[9];
  const float* fcb    = (const float*)d_in[10];
  const float* gamma  = (const float*)d_in[11];
  const float* beta   = (const float*)d_in[12];
  float* out = (float*)d_out;
  float* ws  = (float*)d_ws;

  k_zero_stats<<<1, 64, 0, stream>>>(ws);
  k_routing<<<N_, 256, 0, stream>>>(stdx, epochs, c1w, c1b, c2w, c2b, fcw, fcb, ws);
  k_mix<<<N_ * KS, 256, 0, stream>>>(wts, biases, ws);
  dim3 grid(T_ * H_, N_);
  k_conv<<<grid, 224, SMEM_TOTAL, stream>>>(x, out, ws);
  k_finalize<<<1, 32, 0, stream>>>(gamma, beta, ws);
  k_bnrelu<<<8192, 256, 0, stream>>>(out, ws);
}
